// AnswerSelection_26147760898612
// MI455X (gfx1250) — matrix operand order repaired and checked by reading, not compiled, not run on hardware
//
#include <hip/hip_runtime.h>

// ---------------------------------------------------------------------------
// Fused MLP: score[b,o] = relu(A[o,b,:]*W1^T + b1) * W2^T + b2
// A: [5,100000,600] f32, W1: [75,600], b1:[75], W2:[1,75], b2:[1]
// out: [100000,5] f32
//
// Bandwidth-bound (1.2 GB stream vs 45 GFLOP) -> stream A once, keep W1 in
// LDS as pre-swizzled bf16 WMMA B-fragments, fuse both layers + relu + store.
// A per-tile compiler memory barrier keeps the fragment ds_loads inside the
// tile loop (otherwise LICM hoists 760 VGPRs of fragments and spills to
// scratch, which is reloaded every tile -- seen in round-1 asm).
// ---------------------------------------------------------------------------

typedef __attribute__((ext_vector_type(16))) __bf16        v16bf;
typedef __attribute__((ext_vector_type(8)))  float         v8f;
typedef __attribute__((ext_vector_type(8)))  unsigned int  v8u;

#define DIMK   600
#define HID    75
#define NT     5          // n-tiles of 16 (80 cols, 75 valid)
#define KC     19         // k-chunks of 32 (608, 600 valid)
#define OPT    5
#define BATCH  100000
#define ROWS   (OPT * BATCH)   // 500000
#define TILES  (ROWS / 16)     // 31250 exact, tiles never cross option boundary
#define W1_U32 (NT * KC * 256) // fragments: [NT*KC][4 d2][32 lanes][2 dwords]

static __device__ __forceinline__ unsigned short bf16_bits(float f) {
    __bf16 h = (__bf16)f;
    return __builtin_bit_cast(unsigned short, h);
}

__global__ __launch_bounds__(256, 4) void
answer_selection_mlp(const float* __restrict__ A,
                     const float* __restrict__ W1,
                     const float* __restrict__ b1,
                     const float* __restrict__ W2,
                     const float* __restrict__ b2,
                     float* __restrict__ out)
{
    // W1 as bf16 B-matrix fragments (97,280 B) + padded b1/W2 (640 B)
    __shared__ unsigned int ldsW1[W1_U32];
    __shared__ float b1p[NT * 16];
    __shared__ float w2p[NT * 16];

    const int tid = threadIdx.x;

    // ---- one-time per block: padded bias / second-layer weights -----------
    if (tid < NT * 16) {
        b1p[tid] = (tid < HID) ? b1[tid] : 0.0f;
        w2p[tid] = (tid < HID) ? W2[tid] : 0.0f;   // zero pad kills n>=75 terms
    }

    // ---- one-time per block: swizzle W1 -> bf16 B-fragments in LDS --------
    // V_WMMA_F32_16X16X32_BF16 B (32x16, K x N) lane layout:
//   lanes 0-15 : N=lane,    vgpr d holds K = 2d,2d+1 (d<4), 8+2d,8+2d+1 (d>=4)
//   lanes 16-31: N=lane-16, vgpr d holds K = 8+2d,8+2d+1 (d<4), 16+2d,16+2d+1 (d>=4)
    // LDS fragment storage: uint2-per-lane, d2 stride 64 dwords -> 4
    // conflict-free ds_load_b64 per fragment per lane.
    for (int i = tid; i < NT * KC * 32 * 8; i += 256) {
        const int d    = i & 7;
        const int lane = (i >> 3) & 31;
        const int rest = i >> 8;          // 0..94
        const int c    = rest % KC;
        const int t    = rest / KC;
        const int n    = t * 16 + (lane & 15);
        const int k    = c * 32 + ((lane < 16) ? 0 : 8) + ((d < 4) ? 2 * d : 8 + 2 * d);
        float f0 = 0.0f, f1 = 0.0f;
        if (n < HID) {
            if (k     < DIMK) f0 = W1[n * DIMK + k];
            if (k + 1 < DIMK) f1 = W1[n * DIMK + k + 1];
        }
        const unsigned int packed =
            ((unsigned int)bf16_bits(f1) << 16) | (unsigned int)bf16_bits(f0);
        const int u32i = (t * KC + c) * 256 + (d >> 1) * 64 + lane * 2 + (d & 1);
        ldsW1[u32i] = packed;
    }
    __syncthreads();

    const int  lane    = tid & 31;
    const int  lhalf   = lane & 15;
    const bool hi_half = lane >= 16;
    const int  nWaves  = (int)gridDim.x * 8;
    const int  gw      = (int)blockIdx.x * 8 + (tid >> 5);
    const float bias2  = b2[0];

    // ---- grid-stride over 16-row tiles ------------------------------------
    for (int T = gw; T < TILES; T += nWaves) {
        // Barrier: forbid hoisting the (tile-invariant) LDS fragment loads
        // out of this loop -- keeps VGPR count low and avoids scratch spills.
        asm volatile("" ::: "memory");

        const int rowbase = T * 16;
        // A-fragment lane owns one row (lanes l and l+16 share row l&15)
        const float* __restrict__ arow =
            A + (size_t)(rowbase + lhalf) * DIMK;

        // prefetch next tile's row into cache hierarchy
        if (T + nWaves < TILES) {
            const float* nxt =
                A + (size_t)((T + nWaves) * 16 + lhalf) * DIMK + (hi_half ? 64 : 0);
            __builtin_prefetch(nxt, 0, 1);
        }

        v8f acc[NT];
        #pragma unroll
        for (int t = 0; t < NT; ++t) acc[t] = (v8f)0.0f;

        #pragma unroll 2
        for (int c = 0; c < KC; ++c) {
            // 16-bit A (16x32) layout: lanes 0-15 K runs [0..7],[16..23];
            // lanes 16-31 K runs [8..15],[24..31]
            const int k0 = c * 32 + (hi_half ? 8 : 0);
            float4 a0 = *(const float4*)(arow + k0);
            float4 a1 = *(const float4*)(arow + k0 + 4);
            float4 a2, a3;
            if (c == KC - 1 && hi_half) {
                // second run would be K=600..607 -> zero pad
                a2 = make_float4(0.f, 0.f, 0.f, 0.f);
                a3 = make_float4(0.f, 0.f, 0.f, 0.f);
            } else {
                a2 = *(const float4*)(arow + k0 + 16);
                a3 = *(const float4*)(arow + k0 + 20);
            }
            v16bf af;
            af[0]  = (__bf16)a0.x; af[1]  = (__bf16)a0.y;
            af[2]  = (__bf16)a0.z; af[3]  = (__bf16)a0.w;
            af[4]  = (__bf16)a1.x; af[5]  = (__bf16)a1.y;
            af[6]  = (__bf16)a1.z; af[7]  = (__bf16)a1.w;
            af[8]  = (__bf16)a2.x; af[9]  = (__bf16)a2.y;
            af[10] = (__bf16)a2.z; af[11] = (__bf16)a2.w;
            af[12] = (__bf16)a3.x; af[13] = (__bf16)a3.y;
            af[14] = (__bf16)a3.z; af[15] = (__bf16)a3.w;

            #pragma unroll
            for (int t = 0; t < NT; ++t) {
                const uint2* bp =
                    (const uint2*)(ldsW1 + (t * KC + c) * 256) + lane;
                uint2 q0 = bp[0];
                uint2 q1 = bp[32];
                uint2 q2 = bp[64];
                uint2 q3 = bp[96];
                v8u bu = {q0.x, q0.y, q1.x, q1.y, q2.x, q2.y, q3.x, q3.y};
                v16bf bfrag = __builtin_bit_cast(v16bf, bu);
                acc[t] = __builtin_amdgcn_wmma_f32_16x16x32_bf16(
                    /*neg_a=*/false, af, /*neg_b=*/false, bfrag,
                    /*c_mod=*/(short)0, acc[t],
                    /*reuse_a=*/false, /*reuse_b=*/false);
            }
        }

        // ---- fused layer 2: relu, scale by W2, half-wave reduce, store ----
        // C/D layout: lane l holds N = l&15; vgpr j holds M = j (+8 if l>=16)
        float bb[NT], ww[NT];
        #pragma unroll
        for (int t = 0; t < NT; ++t) {
            bb[t] = b1p[t * 16 + lhalf];
            ww[t] = w2p[t * 16 + lhalf];
        }

        #pragma unroll
        for (int j = 0; j < 8; ++j) {
            float s = 0.0f;
            #pragma unroll
            for (int t = 0; t < NT; ++t)
                s += fmaxf(acc[t][j] + bb[t], 0.0f) * ww[t];
            // reduce over the 16 lanes of this half-wave (N dimension)
            s += __shfl_xor(s, 1, 32);
            s += __shfl_xor(s, 2, 32);
            s += __shfl_xor(s, 4, 32);
            s += __shfl_xor(s, 8, 32);
            if (lhalf == j) {
                const int row = rowbase + j + (hi_half ? 8 : 0); // = o*BATCH + b
                const int o   = row / BATCH;
                const int b   = row - o * BATCH;
                out[(size_t)b * OPT + o] = s + bias2;
            }
        }
    }
}

extern "C" void kernel_launch(void* const* d_in, const int* in_sizes, int n_in,
                              void* d_out, int out_size, void* d_ws, size_t ws_size,
                              hipStream_t stream) {
    (void)in_sizes; (void)n_in; (void)out_size; (void)d_ws; (void)ws_size;
    const float* A  = (const float*)d_in[0];
    const float* W1 = (const float*)d_in[1];
    const float* b1 = (const float*)d_in[2];
    const float* W2 = (const float*)d_in[3];
    const float* b2 = (const float*)d_in[4];
    float* out = (float*)d_out;

    const int blocks = 1024;   // 8192 wave32s, grid-stride over 31250 tiles
    answer_selection_mlp<<<blocks, 256, 0, stream>>>(A, W1, b1, W2, b2, out);
}
